// RnnDecoder_14559939133890
// MI455X (gfx1250) — matrix operand order repaired and checked by reading, not compiled, not run on hardware
//
#include <hip/hip_runtime.h>

#define B_SZ    64
#define L_MAX   2048
#define D_H     512
#define NWG     8
#define THREADS 256

typedef __attribute__((ext_vector_type(16))) __bf16 v16bf;
typedef __attribute__((ext_vector_type(8)))  float  v8f;

// ---- LDS byte layout (dynamic shared) ------------------------------------
// B region: 16 col-tiles * 16 K-chunks * 32 lanes * 32B = 256 KB  (W_hh^T slice, swizzled bf16)
// A region: 16 K-chunks * 32 lanes * 32B               = 16 KB   (h_{t-1}, swizzled bf16)
#define B_OFF   0
#define A_OFF   (16 * 16 * 32 * 32)
#define CNT_OFF (A_OFF + 16 * 32 * 32)
#define SMEM_BYTES (CNT_OFF + 64)

__device__ __forceinline__ int b_off(int nt, int kc, int l) {
    return B_OFF + (((nt * 16 + kc) * 32 + l) << 5);
}
__device__ __forceinline__ int a_off(int kc, int l) {
    return A_OFF + ((kc * 32 + l) << 5);
}

// ---- Kernel 1: u = x @ W_ih^T + b_ih + b_hh; reset grid-barrier counter ---
__global__ void rnn_u_kernel(const float* __restrict__ x, const float* __restrict__ W_ih,
                             const float* __restrict__ b_ih, const float* __restrict__ b_hh,
                             float* __restrict__ u, unsigned* __restrict__ counter) {
    __shared__ float xs[D_H];
    const int b = blockIdx.x, tid = threadIdx.x;
    if (b == 0 && tid == 0) *counter = 0u;
    xs[tid]       = x[b * D_H + tid];
    xs[tid + 256] = x[b * D_H + tid + 256];
    __syncthreads();
    for (int d = tid; d < D_H; d += THREADS) {
        float acc = b_ih[d] + b_hh[d];
        const float* w = W_ih + d * D_H;
        #pragma unroll 8
        for (int k = 0; k < D_H; ++k) acc = fmaf(xs[k], w[k], acc);
        u[b * D_H + d] = acc;
    }
}

// ---- Kernel 2: persistent RNN. 8 WGs = 4 batch-groups x 2 D-halves --------
__global__ void __launch_bounds__(THREADS, 1)
rnn_steps_kernel(const int* __restrict__ counts, const float* __restrict__ W_hh,
                 const float* __restrict__ u, float* __restrict__ out,
                 unsigned short* __restrict__ hbuf, unsigned* __restrict__ counter) {
    extern __shared__ char smem[];
    const int tid  = threadIdx.x, lane = tid & 31, wv = tid >> 5;
    const int group = blockIdx.x >> 1;       // 16 batch rows per group
    const int half  = blockIdx.x & 1;        // 256 hidden columns per half
    int* cnts = (int*)(smem + CNT_OFF);

    if (tid < 16) cnts[tid] = counts[group * 16 + tid];

    // Stage B = W_hh^T slice [K=512, N=256] as bf16 in ISA WMMA-B swizzle:
// lane = n%16 + 16*((kk>>3)&1); dword slot v = (kk>=16?4:0)+((kk&7)>>1); half p = kk&1.
    for (int idx = tid; idx < 256 * D_H; idx += THREADS) {
        const int r = idx >> 9;              // local column n-index 0..255
        const int k = idx & 511;             // K index (coalesced over threads)
        const float w = W_hh[(half * 256 + r) * D_H + k];
        const int nt = r >> 4, n = r & 15;
        const int kc = k >> 5, kk = k & 31;
        const int lh = (kk >> 3) & 1, v = ((kk >= 16) ? 4 : 0) + ((kk & 7) >> 1), p = kk & 1;
        *(__bf16*)(smem + b_off(nt, kc, n + (lh << 4)) + v * 4 + p * 2) = (__bf16)w;
    }
    // h_{-1} = 0
    for (int idx = tid; idx < ((16 * 32 * 32) >> 2); idx += THREADS)
        ((unsigned*)(smem + A_OFF))[idx] = 0u;
    __syncthreads();

    // Wave tile assignment: wave wv owns col-tiles nt0, nt1 (32 columns).
    const int nt0 = wv * 2, nt1 = nt0 + 1;
    const int nloc = lane & 15, mhi = (lane >> 4) << 3;   // C/D layout: n=lane%16, m=v+8*(lane>=16)
    const int d0 = half * 256 + nt0 * 16 + nloc;
    const int d1 = d0 + 16;

    v8f u0, u1;
    int  cnt[8];
    long rowbase[8];
    #pragma unroll
    for (int v = 0; v < 8; ++v) {
        const int m = mhi + v;
        const int brow = group * 16 + m;
        u0[v] = u[brow * D_H + d0];
        u1[v] = u[brow * D_H + d1];
        cnt[v] = cnts[m];
        rowbase[v] = (long)brow * L_MAX * D_H;
    }

    // Own-h scatter constants (A-swizzle of column d: kk=d&31;
    // lane-half lh=(kk>>3)&1; dword slot = (kk>=16?4:0)+((kk&7)>>1); p=kk&1).
    const int kcA0 = d0 >> 5, kkA0 = d0 & 31;
    const int lhA0 = (kkA0 >> 3) & 1, svA0 = ((kkA0 >= 16) ? 4 : 0) + ((kkA0 & 7) >> 1), pA0 = kkA0 & 1;
    const int kcA1 = d1 >> 5, kkA1 = d1 & 31;
    const int lhA1 = (kkA1 >> 3) & 1, svA1 = ((kkA1 >= 16) ? 4 : 0) + ((kkA1 & 7) >> 1), pA1 = kkA1 & 1;

    for (int t = 0; t < L_MAX; ++t) {
        // ---- GEMM: c = u + h_{t-1} @ W_hh^T (slice), bf16 WMMA, f32 accum
        v8f c0 = u0, c1 = u1;
        #pragma unroll
        for (int kc = 0; kc < 16; ++kc) {
            v16bf a  = *(const v16bf*)(smem + a_off(kc, lane));
            v16bf b0 = *(const v16bf*)(smem + b_off(nt0, kc, lane));
            v16bf b1 = *(const v16bf*)(smem + b_off(nt1, kc, lane));
            c0 = __builtin_amdgcn_wmma_f32_16x16x32_bf16(false, a, false, b0, (short)0, c0, false, false);
            c1 = __builtin_amdgcn_wmma_f32_16x16x32_bf16(false, a, false, b1, (short)0, c1, false, false);
        }
        float h0[8], h1[8];
        #pragma unroll
        for (int v = 0; v < 8; ++v) { h0[v] = tanhf(c0[v]); h1[v] = tanhf(c1[v]); }

        // ---- Masked streaming output for step t (zero beyond ragged count)
        const long toff = (long)t * D_H;
        #pragma unroll
        for (int v = 0; v < 8; ++v) {
            const bool live = t < cnt[v];
            __builtin_nontemporal_store(live ? h0[v] : 0.0f, out + rowbase[v] + toff + d0);
            __builtin_nontemporal_store(live ? h1[v] : 0.0f, out + rowbase[v] + toff + d1);
        }

        __syncthreads();                      // everyone done reading A(h_{t-1})

        // ---- Scatter own bf16 h_t into swizzled A region
        #pragma unroll
        for (int v = 0; v < 8; ++v) {
            const int m = mhi + v;
            *(__bf16*)(smem + a_off(kcA0, m + (lhA0 << 4)) + svA0 * 4 + pA0 * 2) = (__bf16)h0[v];
            *(__bf16*)(smem + a_off(kcA1, m + (lhA1 << 4)) + svA1 * 4 + pA1 * 2) = (__bf16)h1[v];
        }
        __syncthreads();                      // own A half complete

        // ---- Export own half (8 KB bf16) to parity buffer in L2
        const int  par = t & 1;
        const long own_base = ((((long)par * 4 + group) * 2 + half) << 12);   // *4096 elems
        #pragma unroll
        for (int q = 0; q < 2; ++q) {
            const int idx8 = tid + q * THREADS;            // 512 groups of 8 elems
            const int m = idx8 >> 5, c8 = (idx8 & 31) << 3;
            const int d = half * 256 + c8;
            const int kc = d >> 5, kk8 = d & 31;
            const int lh = (kk8 >> 3) & 1, vb = (kk8 >= 16) ? 4 : 0;
            uint4 val = *(const uint4*)(smem + a_off(kc, m + (lh << 4)) + vb * 4);
            *(uint4*)(hbuf + own_base + m * 256 + c8) = val;
        }

        // ---- Grid barrier across the 8 persistent workgroups
        __threadfence();
        __syncthreads();
        if (tid == 0) {
            __hip_atomic_fetch_add(counter, 1u, __ATOMIC_RELEASE, __HIP_MEMORY_SCOPE_AGENT);
            const unsigned target = (unsigned)(t + 1) * NWG;
            while (__hip_atomic_load(counter, __ATOMIC_ACQUIRE, __HIP_MEMORY_SCOPE_AGENT) < target)
                __builtin_amdgcn_s_sleep(1);
        }
        __syncthreads();

        // ---- Import partner half into swizzled A
        const long par_base = ((((long)par * 4 + group) * 2 + (half ^ 1)) << 12);
        #pragma unroll
        for (int q = 0; q < 2; ++q) {
            const int idx8 = tid + q * THREADS;
            const int m = idx8 >> 5, c8 = (idx8 & 31) << 3;
            const int d = (half ^ 1) * 256 + c8;
            const int kc = d >> 5, kk8 = d & 31;
            const int lh = (kk8 >> 3) & 1, vb = (kk8 >= 16) ? 4 : 0;
            uint4 val = *(const uint4*)(hbuf + par_base + m * 256 + c8);
            *(uint4*)(smem + a_off(kc, m + (lh << 4)) + vb * 4) = val;
        }
        __syncthreads();
    }
}

extern "C" void kernel_launch(void* const* d_in, const int* in_sizes, int n_in,
                              void* d_out, int out_size, void* d_ws, size_t ws_size,
                              hipStream_t stream) {
    (void)in_sizes; (void)n_in; (void)out_size; (void)ws_size;
    const int*   counts = (const int*)  d_in[0];
    const float* x      = (const float*)d_in[1];
    const float* W_ih   = (const float*)d_in[2];
    const float* W_hh   = (const float*)d_in[3];
    const float* b_ih   = (const float*)d_in[4];
    const float* b_hh   = (const float*)d_in[5];
    float* out = (float*)d_out;

    // Workspace layout: u [64*512 f32] | hbuf [2*4*2*16*256 bf16] | counter
    float*          u       = (float*)d_ws;
    unsigned short* hbuf    = (unsigned short*)((char*)d_ws + B_SZ * D_H * sizeof(float));
    unsigned*       counter = (unsigned*)((char*)hbuf + 2 * 4 * 2 * 16 * 256 * sizeof(unsigned short));

    hipFuncSetAttribute((const void*)rnn_steps_kernel,
                        hipFuncAttributeMaxDynamicSharedMemorySize, SMEM_BYTES);

    rnn_u_kernel<<<B_SZ, THREADS, 0, stream>>>(x, W_ih, b_ih, b_hh, u, counter);
    rnn_steps_kernel<<<NWG, THREADS, SMEM_BYTES, stream>>>(counts, W_hh, u, out, hbuf, counter);
}
